// GCN_34342558499559
// MI455X (gfx1250) — matrix operand order repaired and checked by reading, not compiled, not run on hardware
//
#include <hip/hip_runtime.h>
#include <hip/hip_bf16.h>
#include <math.h>

typedef __attribute__((ext_vector_type(16))) _Float16 v16h;
typedef __attribute__((ext_vector_type(8)))  float    v8f;
typedef __attribute__((ext_vector_type(4)))  int      v4i;

#define GLOBAL_AS __attribute__((address_space(1)))
#define LDS_AS    __attribute__((address_space(3)))

// ---------------------------------------------------------------------------
// CDNA5 async memory->LDS copy (ASYNCcnt-tracked), with portable fallback.
// Builtin signature (per hipcc diagnostic): arg0 = global int4*, arg1 = LDS ptr.
// ---------------------------------------------------------------------------
__device__ __forceinline__ void async_copy_b128(void* lds_dst, const void* g_src) {
#if defined(__AMDGCN__) && __has_builtin(__builtin_amdgcn_global_load_async_to_lds_b128)
  __builtin_amdgcn_global_load_async_to_lds_b128(
      (GLOBAL_AS v4i*)g_src, (LDS_AS v4i*)lds_dst, /*offset=*/0, /*cpol=*/0);
#else
  *(float4*)lds_dst = *(const float4*)g_src;
#endif
}

__device__ __forceinline__ void wait_async_zero() {
#if defined(__AMDGCN__)
#if __has_builtin(__builtin_amdgcn_s_wait_asynccnt)
  __builtin_amdgcn_s_wait_asynccnt(0);
#else
  asm volatile("s_wait_asynccnt 0x0" ::: "memory");
#endif
#endif
}

// ---------------------------------------------------------------------------
// WMMA GEMM: Z[N x HDIM] = X[N x 128] * W[128 x HDIM]   (f32 in, f16 WMMA, f32 out)
// Block: HDIM/16 waves; each wave owns one 16-wide column tile, holds its four
// B fragments in registers, and iterates over RT row tiles staged in LDS.
// Fragment layouts used here (A/B k order is contiguous, not the ISA table order):
//   A 16x32 f16: lane -> row M=lane&15, 16 consecutive K (lo lanes K0..15, hi K16..31)
//   B 32x16 f16: lane -> column N=lane&15, 16 consecutive K (lo lanes K0..15, hi K16..31)
//   D 16x16 f32: elem r -> M = r + 8*(lane>=16), N = lane&15
// ---------------------------------------------------------------------------
template <int HDIM, int RT>
__global__ __launch_bounds__(32 * (HDIM / 16)) void gemm_wmma_f16(
    const float* __restrict__ X, const float* __restrict__ W,
    float* __restrict__ Z, int nrows, int ntiles) {
  constexpr int KD = 128;          // reduction dim (both layers)
  constexpr int NW = HDIM / 16;    // waves per block
  constexpr int NT = NW * 32;

  __shared__ unsigned int Wlds[HDIM][KD / 2];       // transposed: [col n][k-pair]
  __shared__ unsigned int Alds[RT * 16][KD / 2];    // [row][k-pair]

  const int tid = threadIdx.x;
  const int t0  = blockIdx.x * RT;                  // first row tile of block

  // Stage W transposed into LDS as packed f16 pairs (once per RT*16 rows).
  for (int i = tid; i < HDIM * (KD / 2); i += NT) {
    const int n = i / (KD / 2);
    const int p = i % (KD / 2);
    union { _Float16 h[2]; unsigned int u; } pk;
    pk.h[0] = (_Float16)W[(2 * p + 0) * HDIM + n];
    pk.h[1] = (_Float16)W[(2 * p + 1) * HDIM + n];
    Wlds[n][p] = pk.u;
  }
  // Stage RT 16-row A tiles (float4 global loads -> packed f16 pairs).
  for (int i = tid; i < RT * 16 * (KD / 4); i += NT) {
    const int r = i / (KD / 4);
    const int q = i % (KD / 4);                     // float4 index within row
    const int g = t0 * 16 + r;                      // global row
    float4 v = make_float4(0.f, 0.f, 0.f, 0.f);
    if (g < nrows) v = *(const float4*)(X + (size_t)g * KD + q * 4);
    union { _Float16 h[2]; unsigned int u; } p0, p1;
    p0.h[0] = (_Float16)v.x; p0.h[1] = (_Float16)v.y;
    p1.h[0] = (_Float16)v.z; p1.h[1] = (_Float16)v.w;
    Alds[r][2 * q + 0] = p0.u;
    Alds[r][2 * q + 1] = p1.u;
  }
  __syncthreads();

  const int lane = tid & 31;
  const int wave = tid >> 5;
  const int row  = lane & 15;      // A-row (M) within tile this lane feeds
  const int hi   = lane >> 4;      // 0 => low-K half, 1 => high-K half
  const int n0   = wave * 16;      // column tile base
  const int ncol = lane & 15;      // B column / D column this lane feeds

  // Hoist the four B fragments (K chunks of 32) into registers.
  union Frag { v16h v; unsigned int u[8]; };
  Frag bfr[KD / 32];
  #pragma unroll
  for (int kk = 0; kk < KD / 32; ++kk) {
    const int bbase = kk * 16 + hi * 8;
    #pragma unroll
    for (int i = 0; i < 8; ++i) bfr[kk].u[i] = Wlds[n0 + ncol][bbase + i];
  }

  // Sweep row tiles: A-frag ds loads + 4 chained WMMAs + store each.
  for (int it = 0; it < RT; ++it) {
    const int tile = t0 + it;
    if (tile >= ntiles) break;                      // uniform across block
    v8f acc = {};
    #pragma unroll
    for (int kk = 0; kk < KD / 32; ++kk) {
      Frag a;
      const int abase = kk * 16 + hi * 8;
      #pragma unroll
      for (int i = 0; i < 4; ++i) a.u[i]     = Alds[it * 16 + row][abase + i];
      #pragma unroll
      for (int i = 0; i < 4; ++i) a.u[4 + i] = Alds[it * 16 + row][abase + 4 + i];
      acc = __builtin_amdgcn_wmma_f32_16x16x32_f16(
          /*neg_a=*/false, a.v, /*neg_b=*/false, bfr[kk].v,
          /*c_mod=*/(short)0, acc, /*reuse_a=*/false, /*reuse_b=*/false);
    }
    const size_t rbase = (size_t)tile * 16;
    #pragma unroll
    for (int r = 0; r < 8; ++r) {
      const int m = r + hi * 8;
      Z[(rbase + m) * HDIM + n0 + ncol] = acc[r];
    }
  }
}

// ---------------------------------------------------------------------------
// Soft-median aggregation: one block (256 threads) per node.
//   m[k][d] = Z[src[n*16+k]][d]           (gather via async-to-LDS, L2-resident)
//   med[d]  = median_k m[k][d]            (mean of 8th/9th order stats, K=16)
//   dist[k] = || m[k] - med ||_2
//   alpha   = softmax(-dist / sqrt(D))    (T = 1)
//   out[d]  = sum_k alpha[k] m[k][d] + b[d]   (optional relu)
// ---------------------------------------------------------------------------
template <int D, bool RELU>
__global__ __launch_bounds__(256) void softmedian_agg(
    const float* __restrict__ Z, const int* __restrict__ src,
    const float* __restrict__ bias, float* __restrict__ OUT, int nnodes) {
  __shared__ float m[16][D];
  __shared__ float med[D];
  __shared__ float part[16][16];
  __shared__ float dist[16];
  __shared__ float alpha[16];
  __shared__ int   s_src[16];

  const int tid = threadIdx.x;

  for (int n = blockIdx.x; n < nnodes; n += gridDim.x) {
    if (tid < 16) s_src[tid] = src[n * 16 + tid];
    __syncthreads();

    // Gather 16 neighbor rows straight into LDS (ASYNCcnt path on CDNA5).
    constexpr int V = D / 4;                      // float4 per row
    for (int i = tid; i < 16 * V; i += 256) {
      const int k = i / V;
      const int c = i % V;
      async_copy_b128(&m[k][c * 4], Z + (size_t)s_src[k] * D + c * 4);
    }
    wait_async_zero();
    __syncthreads();

    // Per-dimension median of 16 values: fully unrolled compare-swap network
    // (static indices -> stays in VGPRs as v_min/v_max chains).
    if (tid < D) {
      float v[16];
      #pragma unroll
      for (int k = 0; k < 16; ++k) v[k] = m[k][tid];
      #pragma unroll
      for (int i = 0; i < 15; ++i) {
        #pragma unroll
        for (int j = 0; j < 15 - i; ++j) {
          const float lo = fminf(v[j], v[j + 1]);
          const float hh = fmaxf(v[j], v[j + 1]);
          v[j] = lo;
          v[j + 1] = hh;
        }
      }
      med[tid] = 0.5f * (v[7] + v[8]);
    }
    __syncthreads();

    // Distance partials: thread t -> (k = t&15, part p = t>>4 covering D/16 dims).
    {
      const int k = tid & 15;
      const int p = tid >> 4;
      constexpr int DP = D / 16;
      float s = 0.0f;
      #pragma unroll
      for (int dd = 0; dd < DP; ++dd) {
        const int d = p * DP + dd;
        const float t = m[k][d] - med[d];
        s += t * t;
      }
      part[k][p] = s;
    }
    __syncthreads();

    if (tid < 16) {
      float s = 0.0f;
      #pragma unroll
      for (int p = 0; p < 16; ++p) s += part[tid][p];
      dist[tid] = sqrtf(s);
    }
    __syncthreads();

    // Softmax over 16 scores (single lane; trivial work).
    if (tid == 0) {
      const float scale = -1.0f / sqrtf((float)D);   // T = 1
      float sc[16];
      float mx = -3.0e38f;
      #pragma unroll
      for (int k = 0; k < 16; ++k) {
        sc[k] = dist[k] * scale;
        mx = fmaxf(mx, sc[k]);
      }
      float sum = 0.0f;
      #pragma unroll
      for (int k = 0; k < 16; ++k) {
        sc[k] = __expf(sc[k] - mx);
        sum += sc[k];
      }
      const float r = 1.0f / sum;
      #pragma unroll
      for (int k = 0; k < 16; ++k) alpha[k] = sc[k] * r;
    }
    __syncthreads();

    // Weighted sum + bias (+relu).
    if (tid < D) {
      float acc = bias[tid];
      #pragma unroll
      for (int k = 0; k < 16; ++k) acc += alpha[k] * m[k][tid];
      if (RELU) acc = fmaxf(acc, 0.0f);
      OUT[(size_t)n * D + tid] = acc;
    }
    __syncthreads();   // protect LDS reuse on grid-stride
  }
}

// ---------------------------------------------------------------------------
// Inputs (setup_inputs order): x[N*128] f32, src[N*16] i32, W1[128*128] f32,
// b1[128] f32, W2[128*64] f32, b2[64] f32.  Output: [N*64] f32.
// Workspace: z1 (N*128 f32) | h1 (N*128 f32); z2 reuses the z1 region.
// ---------------------------------------------------------------------------
extern "C" void kernel_launch(void* const* d_in, const int* in_sizes, int n_in,
                              void* d_out, int out_size, void* d_ws, size_t ws_size,
                              hipStream_t stream) {
  const float* x   = (const float*)d_in[0];
  const int*   src = (const int*)d_in[1];
  const float* W1  = (const float*)d_in[2];
  const float* b1  = (const float*)d_in[3];
  const float* W2  = (const float*)d_in[4];
  const float* b2  = (const float*)d_in[5];
  float* out = (float*)d_out;

  const int N = in_sizes[0] / 128;          // 50000
  const int ntiles = N / 16;                // 3125 (N divisible by 16)
  constexpr int RT = 8;                     // row tiles per GEMM block
  const int gblocks = (ntiles + RT - 1) / RT;

  float* z1 = (float*)d_ws;                 // N*128
  float* h1 = z1 + (size_t)N * 128;         // N*128
  float* z2 = z1;                           // reuse once z1 is consumed

  // Layer 1: z1 = x @ W1 ; h1 = relu(softmedian(z1) + b1)
  gemm_wmma_f16<128, RT><<<gblocks, 256, 0, stream>>>(x, W1, z1, N, ntiles);
  softmedian_agg<128, true><<<N, 256, 0, stream>>>(z1, src, b1, h1, N);

  // Layer 2: z2 = h1 @ W2 ; out = softmedian(z2) + b2
  gemm_wmma_f16<64, RT><<<gblocks, 128, 0, stream>>>(h1, W2, z2, N, ntiles);
  softmedian_agg<64, false><<<N, 256, 0, stream>>>(z2, src, b2, out, N);
}
